// SimpleAttention_67903432949799
// MI455X (gfx1250) — compile-verified, matrix operand order checked by reading, not run on hardware
//
#include <hip/hip_runtime.h>
#include <hip/hip_bf16.h>
#include <math.h>

// ---------------------------------------------------------------------------
// CDNA5 (gfx1250) bf16 WMMA attention pipeline, round 2.
// All matmuls use v_wmma_f32_16x16x32_bf16 (f32 accumulate).
// GEMMs: 32x64 tile per wave (2 A-frags x 4 B-frags, 8 WMMA/k-step).
// Attention: 32 queries per wave, K/V fragments shared across 2 query blocks.
// ---------------------------------------------------------------------------

typedef __attribute__((ext_vector_type(16))) __bf16 bf16x16;
typedef __attribute__((ext_vector_type(8)))  float  f32x8;

union FragU { bf16x16 v; uint32_t u[8]; };

__device__ __forceinline__ f32x8 zero8() {
  f32x8 z;
#pragma unroll
  for (int i = 0; i < 8; ++i) z[i] = 0.0f;
  return z;
}

// Load a 16x32 bf16 tile in the WMMA A-fragment layout (ISA 7.12.2).
// `base` points at tile origin (row0, k0) of a row-major matrix with leading
// dimension `ld` elements. Also used for B-fragments when B is stored
// N-major (transposed). Per-lane the 4 dword pairs per half are contiguous,
// so the compiler fuses them into global_load_b128 pairs.
__device__ __forceinline__ bf16x16 load_frag(const __bf16* __restrict__ base,
                                             int ld, int lane) {
  const int r  = lane & 15;
  const int hi = (lane >> 4) & 1;
  const __bf16* rowp = base + (size_t)r * ld;
  FragU f;
#pragma unroll
  for (int v = 0; v < 8; ++v) {
    const int kk = ((v & 4) << 2) + (hi << 3) + ((v & 3) << 1);  // K pair index
    f.u[v] = *reinterpret_cast<const uint32_t*>(rowp + kk);      // 2 x bf16
  }
  return f.v;
}

__device__ __forceinline__ f32x8 wmma_bf16(bf16x16 a, bf16x16 b, f32x8 c) {
  // (neg_a, A, neg_b, B, c_mod, C, reuse_a, reuse_b)
  return __builtin_amdgcn_wmma_f32_16x16x32_bf16(false, a, false, b,
                                                 (short)0, c, false, false);
}

// Reductions across the 16 lanes of each wave half (columns of one C row).
__device__ __forceinline__ float rowmax16(float v) {
#pragma unroll
  for (int m = 1; m <= 8; m <<= 1) v = fmaxf(v, __shfl_xor(v, m, 32));
  return v;
}
__device__ __forceinline__ float rowsum16(float v) {
#pragma unroll
  for (int m = 1; m <= 8; m <<= 1) v += __shfl_xor(v, m, 32);
  return v;
}

// ---------------------------------------------------------------------------
// Problem constants
// ---------------------------------------------------------------------------
#define BATCH   2
#define SEQ     2048
#define DIM     1024
#define HEADS   16
#define DHEAD   64
#define EQKV    3072              // 3 * HEADS * DHEAD
#define ROWS    (BATCH * SEQ)     // 4096

// ---------------------------------------------------------------------------
// Kernel 0: fp32 -> bf16 cast (grid-stride)
// ---------------------------------------------------------------------------
__global__ void cast_f32_to_bf16(const float* __restrict__ src,
                                 __bf16* __restrict__ dst, int n) {
  int i = blockIdx.x * blockDim.x + threadIdx.x;
  const int stride = gridDim.x * blockDim.x;
  for (; i < n; i += stride) dst[i] = (__bf16)src[i];
}

// ---------------------------------------------------------------------------
// Kernel 1: QKV projection.  qkv[m][e] = sum_d Xb[m][d] * Wqkv[e][d]
// 32(M) x 64(N) tile per wave; B-fragments reused across both A-fragments.
// Scatter: e = h*192 + t;  t<64 -> Q (pre-scaled by 1/8), t<128 -> K,
// else -> V stored transposed [bh][dh][n] for packed PV B-loads.
// ---------------------------------------------------------------------------
__global__ __launch_bounds__(32)
void qkv_gemm_kernel(const __bf16* __restrict__ Xb,
                     const __bf16* __restrict__ Wb,
                     __bf16* __restrict__ Q,
                     __bf16* __restrict__ K,
                     __bf16* __restrict__ Vt) {
  const int lane = threadIdx.x;
  const int m0 = blockIdx.x * 32;
  const int n0 = blockIdx.y * 64;
  const int r  = lane & 15;
  const int hi = (lane >> 4) & 1;

  f32x8 acc[2][4];
#pragma unroll
  for (int q = 0; q < 2; ++q)
#pragma unroll
    for (int j = 0; j < 4; ++j) acc[q][j] = zero8();

  for (int k0 = 0; k0 < DIM; k0 += 32) {
    const __bf16* aBase = Xb + (size_t)m0 * DIM + k0;
    __builtin_prefetch(aBase + 32, 0, 3);            // global_prefetch_b8
    bf16x16 a0 = load_frag(aBase,             DIM, lane);
    bf16x16 a1 = load_frag(aBase + 16 * DIM,  DIM, lane);
#pragma unroll
    for (int j = 0; j < 4; ++j) {
      bf16x16 b = load_frag(Wb + (size_t)(n0 + j * 16) * DIM + k0, DIM, lane);
      acc[0][j] = wmma_bf16(a0, b, acc[0][j]);
      acc[1][j] = wmma_bf16(a1, b, acc[1][j]);
    }
  }

#pragma unroll
  for (int q = 0; q < 2; ++q) {
#pragma unroll
    for (int j = 0; j < 4; ++j) {
      const int e = n0 + j * 16 + r;
      const int h = e / 192;
      const int t = e % 192;
#pragma unroll
      for (int i = 0; i < 8; ++i) {
        const int m  = m0 + q * 16 + i + hi * 8;
        const int b  = m >> 11;        // / SEQ
        const int nq = m & (SEQ - 1);
        const int bh = b * HEADS + h;
        const float v = acc[q][j][i];
        if (t < 64) {
          Q[((size_t)bh * SEQ + nq) * DHEAD + t] = (__bf16)(v * 0.125f);
        } else if (t < 128) {
          K[((size_t)bh * SEQ + nq) * DHEAD + (t - 64)] = (__bf16)v;
        } else {
          Vt[((size_t)bh * DHEAD + (t - 128)) * SEQ + nq] = (__bf16)v;
        }
      }
    }
  }
}

// ---------------------------------------------------------------------------
// Online-softmax update for one 16-row query block against 32 fresh keys.
// C-layout: VGPR i -> row i+8*hi, lane -> column. Writes bf16 P rows to LDS.
// ---------------------------------------------------------------------------
__device__ __forceinline__ void softmax_update(const f32x8& sa, const f32x8& sb,
                                               float* __restrict__ mrow,
                                               float* __restrict__ lrow,
                                               f32x8* __restrict__ o,
                                               __bf16* __restrict__ ldsBlock,
                                               int r, int hi) {
#pragma unroll
  for (int i = 0; i < 8; ++i) {
    const float mx   = rowmax16(fmaxf(sa[i], sb[i]));
    const float mnew = fmaxf(mrow[i], mx);
    const float pa = __expf(sa[i] - mnew);
    const float pb = __expf(sb[i] - mnew);
    const float rs = rowsum16(pa + pb);
    const float alpha = __expf(mrow[i] - mnew);
    lrow[i] = lrow[i] * alpha + rs;
    mrow[i] = mnew;
#pragma unroll
    for (int t = 0; t < 4; ++t) o[t][i] *= alpha;
    const int row = i + hi * 8;
    ldsBlock[row * 32 + r]      = (__bf16)pa;
    ldsBlock[row * 32 + 16 + r] = (__bf16)pb;
  }
}

// ---------------------------------------------------------------------------
// Kernel 2: flash attention.  One wave per (bh, 32-query tile).
// Per 32-key step: 8 QK^T WMMAs + 8 PV WMMAs; K/V fragments loaded once and
// shared across both 16-row query blocks (32 flop/byte toward L2).
// ---------------------------------------------------------------------------
__global__ __launch_bounds__(32)
void attn_kernel(const __bf16* __restrict__ Q,
                 const __bf16* __restrict__ K,
                 const __bf16* __restrict__ Vt,
                 __bf16* __restrict__ Ob) {
  __shared__ __bf16 ldsP[32 * 32];    // 32 query rows x 32 keys

  const int lane = threadIdx.x;
  const int qt = blockIdx.x;          // 0..63
  const int bh = blockIdx.y;          // 0..31
  const int r  = lane & 15;
  const int hi = (lane >> 4) & 1;
  const int m0 = qt * 32;

  const __bf16* Qbase = Q  + ((size_t)bh * SEQ + m0) * DHEAD;
  const __bf16* Kbase = K  + (size_t)bh * SEQ * DHEAD;
  const __bf16* Vbase = Vt + (size_t)bh * DHEAD * SEQ;

  const bf16x16 q00 = load_frag(Qbase + 0,               DHEAD, lane); // rows 0-15,  d 0-31
  const bf16x16 q01 = load_frag(Qbase + 32,              DHEAD, lane); // rows 0-15,  d 32-63
  const bf16x16 q10 = load_frag(Qbase + 16 * DHEAD + 0,  DHEAD, lane); // rows 16-31, d 0-31
  const bf16x16 q11 = load_frag(Qbase + 16 * DHEAD + 32, DHEAD, lane); // rows 16-31, d 32-63

  f32x8 o0[4], o1[4];
#pragma unroll
  for (int t = 0; t < 4; ++t) { o0[t] = zero8(); o1[t] = zero8(); }
  float m0row[8], l0row[8], m1row[8], l1row[8];
#pragma unroll
  for (int i = 0; i < 8; ++i) {
    m0row[i] = -INFINITY; l0row[i] = 0.0f;
    m1row[i] = -INFINITY; l1row[i] = 0.0f;
  }

  const f32x8 cz = zero8();

  for (int j0 = 0; j0 < SEQ; j0 += 32) {
    // ---- K fragments for 32 keys (shared by both query blocks) ----
    bf16x16 ka0 = load_frag(Kbase + (size_t)j0 * DHEAD + 0,         DHEAD, lane);
    bf16x16 ka1 = load_frag(Kbase + (size_t)j0 * DHEAD + 32,        DHEAD, lane);
    bf16x16 kb0 = load_frag(Kbase + (size_t)(j0 + 16) * DHEAD + 0,  DHEAD, lane);
    bf16x16 kb1 = load_frag(Kbase + (size_t)(j0 + 16) * DHEAD + 32, DHEAD, lane);

    // ---- scores: S[32 x 32] as four 16x16 tiles (scale folded into Q) ----
    f32x8 s0a = wmma_bf16(q00, ka0, cz);  s0a = wmma_bf16(q01, ka1, s0a);
    f32x8 s0b = wmma_bf16(q00, kb0, cz);  s0b = wmma_bf16(q01, kb1, s0b);
    f32x8 s1a = wmma_bf16(q10, ka0, cz);  s1a = wmma_bf16(q11, ka1, s1a);
    f32x8 s1b = wmma_bf16(q10, kb0, cz);  s1b = wmma_bf16(q11, kb1, s1b);

    // ---- online softmax per query block, P -> LDS ----
    softmax_update(s0a, s0b, m0row, l0row, o0, ldsP,           r, hi);
    softmax_update(s1a, s1b, m1row, l1row, o1, ldsP + 16 * 32, r, hi);
    __syncthreads();

    // ---- P (2 x 16x32) as A-fragments; V^T rows give packed B-fragments ----
    const bf16x16 pA0 = load_frag(ldsP,           32, lane);
    const bf16x16 pA1 = load_frag(ldsP + 16 * 32, 32, lane);
#pragma unroll
    for (int t = 0; t < 4; ++t) {
      bf16x16 vB = load_frag(Vbase + (size_t)(t * 16) * SEQ + j0, SEQ, lane);
      o0[t] = wmma_bf16(pA0, vB, o0[t]);
      o1[t] = wmma_bf16(pA1, vB, o1[t]);
    }
    __syncthreads();
  }

  // ---- normalize and write O as [b, n, h*64 + d] in bf16 ----
  const int b = bh >> 4;
  const int h = bh & 15;
#pragma unroll
  for (int i = 0; i < 8; ++i) {
    const float inv0 = 1.0f / l0row[i];
    const float inv1 = 1.0f / l1row[i];
    const int nq0 = m0 + i + hi * 8;
    const int nq1 = m0 + 16 + i + hi * 8;
    const size_t orow0 = ((size_t)b * SEQ + nq0) * DIM;
    const size_t orow1 = ((size_t)b * SEQ + nq1) * DIM;
#pragma unroll
    for (int t = 0; t < 4; ++t) {
      Ob[orow0 + h * DHEAD + t * 16 + r] = (__bf16)(o0[t][i] * inv0);
      Ob[orow1 + h * DHEAD + t * 16 + r] = (__bf16)(o1[t][i] * inv1);
    }
  }
}

// ---------------------------------------------------------------------------
// Kernel 3: output projection.  out[m][e] = sum_k Ob[m][k]*Wout[e][k] + bout[e]
// 32x64 tile per wave, same blocking as kernel 1.
// ---------------------------------------------------------------------------
__global__ __launch_bounds__(32)
void proj_gemm_kernel(const __bf16* __restrict__ Ob,
                      const __bf16* __restrict__ Wb,
                      const float* __restrict__ bout,
                      float* __restrict__ out) {
  const int lane = threadIdx.x;
  const int m0 = blockIdx.x * 32;
  const int n0 = blockIdx.y * 64;
  const int r  = lane & 15;
  const int hi = (lane >> 4) & 1;

  f32x8 acc[2][4];
#pragma unroll
  for (int q = 0; q < 2; ++q)
#pragma unroll
    for (int j = 0; j < 4; ++j) acc[q][j] = zero8();

  for (int k0 = 0; k0 < DIM; k0 += 32) {
    const __bf16* aBase = Ob + (size_t)m0 * DIM + k0;
    __builtin_prefetch(aBase + 32, 0, 3);
    bf16x16 a0 = load_frag(aBase,            DIM, lane);
    bf16x16 a1 = load_frag(aBase + 16 * DIM, DIM, lane);
#pragma unroll
    for (int j = 0; j < 4; ++j) {
      bf16x16 b = load_frag(Wb + (size_t)(n0 + j * 16) * DIM + k0, DIM, lane);
      acc[0][j] = wmma_bf16(a0, b, acc[0][j]);
      acc[1][j] = wmma_bf16(a1, b, acc[1][j]);
    }
  }

#pragma unroll
  for (int j = 0; j < 4; ++j) {
    const int e = n0 + j * 16 + r;
    const float bias = bout[e];
#pragma unroll
    for (int q = 0; q < 2; ++q) {
#pragma unroll
      for (int i = 0; i < 8; ++i) {
        const int m = m0 + q * 16 + i + hi * 8;
        out[(size_t)m * DIM + e] = acc[q][j][i] + bias;
      }
    }
  }
}

// ---------------------------------------------------------------------------
// Launcher.  Workspace layout (bytes):
//   Xb     @ 0         : 8 MiB     Wqkvb @ 8 MiB  : 6 MiB
//   Woutb  @ 14 MiB    : 2 MiB     Q     @ 16 MiB : 8 MiB (pre-scaled 1/8)
//   K      @ 24 MiB    : 8 MiB     Vt    @ 32 MiB : 8 MiB (transposed/head)
//   Ob     @ 40 MiB    : 8 MiB     -> total 48 MiB
// ---------------------------------------------------------------------------
extern "C" void kernel_launch(void* const* d_in, const int* in_sizes, int n_in,
                              void* d_out, int out_size, void* d_ws, size_t ws_size,
                              hipStream_t stream) {
  (void)in_sizes; (void)n_in; (void)out_size; (void)ws_size;

  const float* x    = (const float*)d_in[0];
  const float* Wqkv = (const float*)d_in[1];
  const float* Wout = (const float*)d_in[2];
  const float* bout = (const float*)d_in[3];
  float* out = (float*)d_out;

  char* ws = (char*)d_ws;
  __bf16* Xb    = (__bf16*)(ws);
  __bf16* Wqkvb = (__bf16*)(ws + (8u  << 20));
  __bf16* Woutb = (__bf16*)(ws + (14u << 20));
  __bf16* Qb    = (__bf16*)(ws + (16u << 20));
  __bf16* Kb    = (__bf16*)(ws + (24u << 20));
  __bf16* Vtb   = (__bf16*)(ws + (32u << 20));
  __bf16* Ob    = (__bf16*)(ws + (40u << 20));

  const int NX = ROWS * DIM;        // 4,194,304
  const int NW = EQKV * DIM;        // 3,145,728
  const int NO = DIM * DIM;         // 1,048,576

  cast_f32_to_bf16<<<4096, 256, 0, stream>>>(x,    Xb,    NX);
  cast_f32_to_bf16<<<4096, 256, 0, stream>>>(Wqkv, Wqkvb, NW);
  cast_f32_to_bf16<<<2048, 256, 0, stream>>>(Wout, Woutb, NO);

  qkv_gemm_kernel<<<dim3(ROWS / 32, EQKV / 64), 32, 0, stream>>>(
      Xb, Wqkvb, Qb, Kb, Vtb);

  attn_kernel<<<dim3(SEQ / 32, BATCH * HEADS), 32, 0, stream>>>(
      Qb, Kb, Vtb, Ob);

  proj_gemm_kernel<<<dim3(ROWS / 32, DIM / 64), 32, 0, stream>>>(
      Ob, Woutb, bout, out);
}
